// Conv2d_STN_15762529976348
// MI455X (gfx1250) — matrix operand order repaired and checked by reading, not compiled, not run on hardware
//
#include <hip/hip_runtime.h>
#include <hip/hip_bf16.h>

typedef _Float16 h16;
typedef __attribute__((ext_vector_type(16))) _Float16 v16h;
typedef __attribute__((ext_vector_type(8)))  _Float16 v8h;
typedef __attribute__((ext_vector_type(8)))  float    v8f;

// ---------------------------------------------------------------------------
// WMMA fragment loaders: K-contiguous 128-bit loads, no per-element math.
//   A 16x32 f16 layout: lane m=lo; k = kk + 16*hi + {0..15}   (fully contiguous)
//   B 32x16 f16 layout: lane n=lo; k = kk + 16*hi + {0..15}   (fully contiguous)
// ---------------------------------------------------------------------------
__device__ __forceinline__ v16h load_a_frag(const h16* __restrict__ rowp, int kk, int hi) {
  return *(const v16h*)(rowp + kk + (hi << 4));          // 32B aligned
}
__device__ __forceinline__ v16h load_b_frag(const h16* __restrict__ rowp, int kk, int hi) {
  return *(const v16h*)(rowp + kk + (hi << 4));          // 32B aligned
}
__device__ __forceinline__ v8f wmma16(v16h a, v16h b, v8f c) {
  return __builtin_amdgcn_wmma_f32_16x16x32_f16(false, a, false, b, (short)0, c, false, false);
}

// ---------------------------------------------------------------------------
// f32 -> f16 cast (optionally zero-padding the tail: used to pad last_w 201->208 rows)
// ---------------------------------------------------------------------------
__global__ void cast_f16_kernel(const float* __restrict__ src, h16* __restrict__ dst,
                                int n, int ntot) {
  int i = blockIdx.x * blockDim.x + threadIdx.x;
  if (i < ntot) dst[i] = (i < n) ? (h16)src[i] : (h16)0.0f;
}

// ---------------------------------------------------------------------------
// Explicit im2col (3x3, pad=1) -> f16 column matrix [B*H*W][2304].
// One thread per (pixel, in-channel); writes 9 contiguous halves.
// ---------------------------------------------------------------------------
__global__ void im2col_kernel(const float* __restrict__ x, h16* __restrict__ col,
                              int H, int W, int Np) {
  int id = blockIdx.x * blockDim.x + threadIdx.x;
  if (id >= Np * 256) return;
  int p = id >> 8, ci = id & 255;
  int HW = H * W;
  int b = p / HW, rem = p % HW, oy = rem / W, ox = rem % W;
  const float* xp = x + ((size_t)(b * 256 + ci)) * HW;
  h16* cp = col + (size_t)p * 2304 + ci * 9;
#pragma unroll
  for (int ky = 0; ky < 3; ++ky) {
    int iy = oy + ky - 1;
#pragma unroll
    for (int kx = 0; kx < 3; ++kx) {
      int ix = ox + kx - 1;
      bool inb = ((unsigned)iy < (unsigned)H) && ((unsigned)ix < (unsigned)W);
      cp[ky * 3 + kx] = inb ? (h16)xp[iy * W + ix] : (h16)0.0f;
    }
  }
}

// ---------------------------------------------------------------------------
// stn0 conv as GEMM: wh[256][2304] x col[Ncols][2304]^T + bias, relu.
// Wave computes 16(M) x 64(N): 4 accumulators share one A fragment.
// Block = 4 waves on 4 consecutive M tiles (B columns shared via cache).
// ---------------------------------------------------------------------------
__global__ __launch_bounds__(128) void conv0_gemm_kernel(
    const h16* __restrict__ wh, const h16* __restrict__ col,
    const float* __restrict__ bias, float* __restrict__ out, int H, int W)
{
  const int lo = threadIdx.x & 15, hi = threadIdx.x >> 4;
  const int mt = blockIdx.x * 4 + threadIdx.y;
  const int nc0 = blockIdx.y * 64;
  const h16* ap = wh + (size_t)(mt * 16 + lo) * 2304;
  const h16* bp0 = col + (size_t)(nc0 + 0 * 16 + lo) * 2304;
  const h16* bp1 = col + (size_t)(nc0 + 1 * 16 + lo) * 2304;
  const h16* bp2 = col + (size_t)(nc0 + 2 * 16 + lo) * 2304;
  const h16* bp3 = col + (size_t)(nc0 + 3 * 16 + lo) * 2304;
  v8f c0 = {}, c1 = {}, c2 = {}, c3 = {};
  for (int kk = 0; kk < 2304; kk += 32) {
    v16h a = load_a_frag(ap, kk, hi);
    c0 = wmma16(a, load_b_frag(bp0, kk, hi), c0);
    c1 = wmma16(a, load_b_frag(bp1, kk, hi), c1);
    c2 = wmma16(a, load_b_frag(bp2, kk, hi), c2);
    c3 = wmma16(a, load_b_frag(bp3, kk, hi), c3);
  }
  const int HW = H * W;
  v8f accs[4] = {c0, c1, c2, c3};
#pragma unroll
  for (int s = 0; s < 4; ++s) {
    int p = nc0 + s * 16 + lo;
    int b = p / HW, rem = p % HW, oy = rem / W, ox = rem % W;
#pragma unroll
    for (int r = 0; r < 8; ++r) {
      int m = mt * 16 + r + (hi << 3);
      float v = accs[s][r] + bias[m];
      out[((b * 256 + m) * H + oy) * W + ox] = v > 0.0f ? v : 0.0f;
    }
  }
}

// ---------------------------------------------------------------------------
// feat GEMM: xs[rows][2304] x conv_w[1024][2304]^T + conv_b, relu -> f16.
// Wave: 16 sample rows x 64 features (4 acc share A).
// ---------------------------------------------------------------------------
__global__ __launch_bounds__(128) void gemm_feat_kernel(
    const h16* __restrict__ xs, const h16* __restrict__ cw,
    const float* __restrict__ cb, h16* __restrict__ fr)
{
  const int lo = threadIdx.x & 15, hi = threadIdx.x >> 4;
  const int mt = blockIdx.x * 4 + threadIdx.y;   // sample tile
  const int fc0 = blockIdx.y * 64;               // feature base
  const h16* ap = xs + (size_t)(mt * 16 + lo) * 2304;
  const h16* bp0 = cw + (size_t)(fc0 + 0 * 16 + lo) * 2304;
  const h16* bp1 = cw + (size_t)(fc0 + 1 * 16 + lo) * 2304;
  const h16* bp2 = cw + (size_t)(fc0 + 2 * 16 + lo) * 2304;
  const h16* bp3 = cw + (size_t)(fc0 + 3 * 16 + lo) * 2304;
  v8f c0 = {}, c1 = {}, c2 = {}, c3 = {};
  for (int kk = 0; kk < 2304; kk += 32) {
    v16h a = load_a_frag(ap, kk, hi);
    c0 = wmma16(a, load_b_frag(bp0, kk, hi), c0);
    c1 = wmma16(a, load_b_frag(bp1, kk, hi), c1);
    c2 = wmma16(a, load_b_frag(bp2, kk, hi), c2);
    c3 = wmma16(a, load_b_frag(bp3, kk, hi), c3);
  }
  v8f accs[4] = {c0, c1, c2, c3};
#pragma unroll
  for (int s = 0; s < 4; ++s) {
    int f = fc0 + s * 16 + lo;
    float bias = cb[f];
#pragma unroll
    for (int r = 0; r < 8; ++r) {
      int n = mt * 16 + r + (hi << 3);
      float v = accs[s][r] + bias;
      fr[(size_t)n * 1024 + f] = (h16)(v > 0.0f ? v : 0.0f);
    }
  }
}

// ---------------------------------------------------------------------------
// class GEMM: lwp[208][1024] (zero-padded) x relu(feat)[rows][1024]^T + last_b.
// Wave: 16 classes x 64 samples (4 acc share A). Block = 2 waves on N.
// ---------------------------------------------------------------------------
__global__ __launch_bounds__(64) void gemm_last_kernel(
    const h16* __restrict__ lwp, const h16* __restrict__ fr,
    const float* __restrict__ lb, float* __restrict__ sbuf,
    int S, int baseoff)
{
  const int lo = threadIdx.x & 15, hi = threadIdx.x >> 4;
  const int mt = blockIdx.x;                               // class tile (0..12)
  const int nc0 = (blockIdx.y * 2 + threadIdx.y) * 64;     // sample base
  const h16* ap = lwp + (size_t)(mt * 16 + lo) * 1024;
  const h16* bp0 = fr + (size_t)(nc0 + 0 * 16 + lo) * 1024;
  const h16* bp1 = fr + (size_t)(nc0 + 1 * 16 + lo) * 1024;
  const h16* bp2 = fr + (size_t)(nc0 + 2 * 16 + lo) * 1024;
  const h16* bp3 = fr + (size_t)(nc0 + 3 * 16 + lo) * 1024;
  v8f c0 = {}, c1 = {}, c2 = {}, c3 = {};
  for (int kk = 0; kk < 1024; kk += 32) {
    v16h a = load_a_frag(ap, kk, hi);
    c0 = wmma16(a, load_b_frag(bp0, kk, hi), c0);
    c1 = wmma16(a, load_b_frag(bp1, kk, hi), c1);
    c2 = wmma16(a, load_b_frag(bp2, kk, hi), c2);
    c3 = wmma16(a, load_b_frag(bp3, kk, hi), c3);
  }
  v8f accs[4] = {c0, c1, c2, c3};
  const int SS = S * S;
#pragma unroll
  for (int s = 0; s < 4; ++s) {
    int n = nc0 + s * 16 + lo;
    int b = n / SS, rem = n % SS;
    int x = rem / S, y = rem % S;
#pragma unroll
    for (int r = 0; r < 8; ++r) {
      int o = mt * 16 + r + (hi << 3);
      if (o < 201)
        sbuf[((size_t)b * 201 + o) * 388 + baseoff + y * S + x] = accs[s][r] + lb[o];
    }
  }
}

// ---------------------------------------------------------------------------
// stn1 conv (256->6, no pad) + identity -> theta; writes theta_diff output and
// the aux sampling coordinates. t = b*S*S + x*S + y.
// ---------------------------------------------------------------------------
__global__ void theta_aux_kernel(
    const float* __restrict__ h, const float* __restrict__ w1,
    const int* __restrict__ checkp, float* __restrict__ tdiff,
    float* __restrict__ auxx, float* __restrict__ auxy,
    int S, int H, int N, int toff)
{
  int t = blockIdx.x * blockDim.x + threadIdx.x;
  if (t >= N) return;
  int SS = S * S;
  int b = t / SS, rem = t % SS;
  int x = rem / S, y = rem % S;
  float fc = 1.0f - (float)checkp[0];
  const float I[6] = {1.f, 0.f, 0.f, 0.f, 1.f, 0.f};
  float th[6];
  for (int o = 0; o < 6; ++o) {
    float acc = 0.0f;
    const float* wp = w1 + o * 2304;
    for (int ci = 0; ci < 256; ++ci) {
      const float* hp = h + ((b * 256 + ci) * H + y) * H + x;   // W == H
      const float* wq = wp + ci * 9;
#pragma unroll
      for (int ky = 0; ky < 3; ++ky)
#pragma unroll
        for (int kx = 0; kx < 3; ++kx)
          acc += hp[ky * H + kx] * wq[ky * 3 + kx];
    }
    th[o] = acc * fc + I[o];
    tdiff[(size_t)(toff + t) * 6 + o] = I[o] - th[o];
  }
  float inv = 2.0f / (float)(H - 1);
#pragma unroll
  for (int i = 0; i < 3; ++i)
#pragma unroll
    for (int j = 0; j < 3; ++j) {
      float bx = (float)(j - 1), by = (float)(i - 1);
      float gx = th[0] * bx + th[1] * by + th[2];
      float gy = th[3] * bx + th[4] * by + th[5];
      auxx[t * 9 + i * 3 + j] = (gx + 1.0f + (float)x) * inv - 1.0f;
      auxy[t * 9 + i * 3 + j] = (gy + 1.0f + (float)y) * inv - 1.0f;
    }
}

// ---------------------------------------------------------------------------
// Bilinear grid-sample -> xs f16 [N][2304]
// ---------------------------------------------------------------------------
__global__ void sample_kernel(
    const float* __restrict__ x, const float* __restrict__ auxx,
    const float* __restrict__ auxy, h16* __restrict__ xs,
    int SS, int H, int Ntot)
{
  int id = blockIdx.x * blockDim.x + threadIdx.x;
  if (id >= Ntot) return;
  int n = id / 9, pos = id % 9;
  int b = n / SS;
  float gx = (auxx[n * 9 + pos] + 1.0f) * 0.5f * (float)(H - 1);
  float gy = (auxy[n * 9 + pos] + 1.0f) * 0.5f * (float)(H - 1);
  float x0f = floorf(gx), y0f = floorf(gy);
  float wx1 = gx - x0f, wy1 = gy - y0f;
  int x0 = (int)x0f, y0 = (int)y0f;
  float w00 = (1.f - wy1) * (1.f - wx1), w01 = (1.f - wy1) * wx1;
  float w10 = wy1 * (1.f - wx1),         w11 = wy1 * wx1;
  bool vx0 = (x0 >= 0) && (x0 < H),     vx1 = (x0 + 1 >= 0) && (x0 + 1 < H);
  bool vy0 = (y0 >= 0) && (y0 < H),     vy1 = (y0 + 1 >= 0) && (y0 + 1 < H);
  int xc0 = min(max(x0, 0), H - 1),     xc1 = min(max(x0 + 1, 0), H - 1);
  int yc0 = min(max(y0, 0), H - 1),     yc1 = min(max(y0 + 1, 0), H - 1);
  const float* xb = x + (size_t)(b * 256) * H * H;
  for (int c = 0; c < 256; ++c) {
    const float* xc = xb + c * H * H;
    float v00 = (vy0 && vx0) ? xc[yc0 * H + xc0] : 0.0f;
    float v01 = (vy0 && vx1) ? xc[yc0 * H + xc1] : 0.0f;
    float v10 = (vy1 && vx0) ? xc[yc1 * H + xc0] : 0.0f;
    float v11 = (vy1 && vx1) ? xc[yc1 * H + xc1] : 0.0f;
    float v = v00 * w00 + v01 * w01 + v10 * w10 + v11 * w11;
    xs[(size_t)n * 2304 + c * 9 + pos] = (h16)v;
  }
}

// ---------------------------------------------------------------------------
// Per-batch softmax statistics over the 201*388 joint score field.
// ---------------------------------------------------------------------------
__global__ void softmax_stats_kernel(const float* __restrict__ sbuf,
                                     float* __restrict__ bmax, float* __restrict__ bsum)
{
  __shared__ float red[256];
  const int TOT = 201 * 388;
  int b = blockIdx.x;
  const float* p = sbuf + (size_t)b * TOT;
  float m = -3.4e38f;
  for (int i = threadIdx.x; i < TOT; i += 256) m = fmaxf(m, p[i]);
  red[threadIdx.x] = m; __syncthreads();
  for (int s = 128; s > 0; s >>= 1) {
    if (threadIdx.x < s) red[threadIdx.x] = fmaxf(red[threadIdx.x], red[threadIdx.x + s]);
    __syncthreads();
  }
  float gm = red[0]; __syncthreads();
  float sum = 0.0f;
  for (int i = threadIdx.x; i < TOT; i += 256) sum += __expf(p[i] - gm);
  red[threadIdx.x] = sum; __syncthreads();
  for (int s = 128; s > 0; s >>= 1) {
    if (threadIdx.x < s) red[threadIdx.x] += red[threadIdx.x + s];
    __syncthreads();
  }
  if (threadIdx.x == 0) { bmax[b] = gm; bsum[b] = red[0]; }
}

__global__ void likelihood_kernel(const float* __restrict__ sbuf,
                                  const float* __restrict__ bmax,
                                  const float* __restrict__ bsum,
                                  float* __restrict__ lik)
{
  int id = blockIdx.x * blockDim.x + threadIdx.x;
  if (id >= 32 * 201) return;
  int b = id / 201;
  const float* p = sbuf + (size_t)id * 388;
  float m = bmax[b], s = 0.0f;
  for (int i = 0; i < 388; ++i) s += __expf(p[i] - m);
  lik[id] = s / bsum[b];
}

// ---------------------------------------------------------------------------
// Boxes / NT boxes / reg_score.
// ---------------------------------------------------------------------------
__device__ void level_box(const float* sr, int base, int L, int S,
                          const float* ax, const float* ay, int b,
                          float m, float Z, int Hdim, float dim,
                          float* box, float* boxNT, float* conf_out)
{
  int pbest = 0; float sbest = sr[base];
  for (int p = 1; p < L; ++p) {
    float v = sr[base + p];
    if (v > sbest) { sbest = v; pbest = p; }
  }
  float conf = __expf(sbest - m) / Z;
  int xsel = pbest % S, ysel = pbest / S;   // p = y*S + x
  int n = b * S * S + xsel * S + ysel;      // aux row
  float txmin = 1e30f, txmax = -1e30f, tymin = 1e30f, tymax = -1e30f;
  for (int i = 0; i < 9; ++i) {
    float tx = (ax[n * 9 + i] + 1.0f) * 0.5f;
    float ty = (ay[n * 9 + i] + 1.0f) * 0.5f;
    txmin = fminf(txmin, tx); txmax = fmaxf(txmax, tx);
    tymin = fminf(tymin, ty); tymax = fmaxf(tymax, ty);
  }
  float lo = 0.0f, hi = dim - 1.0f;
  box[0] = fminf(fmaxf(txmin * dim, lo), hi);
  box[1] = fminf(fmaxf(tymin * dim, lo), hi);
  box[2] = fminf(fmaxf(txmax * dim, lo), hi);
  box[3] = fminf(fmaxf(tymax * dim, lo), hi);
  box[4] = conf;
  float invH = 1.0f / (float)(Hdim - 1);
  boxNT[0] = fminf(fmaxf((float)xsel * invH * dim, lo), hi);
  boxNT[1] = fminf(fmaxf((float)ysel * invH * dim, lo), hi);
  boxNT[2] = fminf(fmaxf((float)(xsel + 2) * invH * dim, lo), hi);
  boxNT[3] = fminf(fmaxf((float)(ysel + 2) * invH * dim, lo), hi);
  boxNT[4] = conf;
  *conf_out = conf;
}

__global__ void final_kernel(const float* __restrict__ lik, const float* __restrict__ sbuf,
                             const float* __restrict__ bmax, const float* __restrict__ bsum,
                             const float* __restrict__ a5x, const float* __restrict__ a5y,
                             const float* __restrict__ a4x, const float* __restrict__ a4y,
                             const int* __restrict__ dimp,
                             float* __restrict__ boxes, float* __restrict__ boxesNT,
                             float* __restrict__ reg)
{
  __shared__ float rr[32];
  int b = threadIdx.x;
  int pred = 0; float bm = lik[b * 201];
  for (int o = 1; o < 201; ++o) {
    float v = lik[b * 201 + o];
    if (v > bm) { bm = v; pred = o; }
  }
  const float* sr = sbuf + ((size_t)b * 201 + pred) * 388;
  float dim = (float)dimp[0];
  float box5[5], bnt5[5], box4[5], bnt4[5], conf5, conf4;
  level_box(sr, 0,  64,  8,  a5x, a5y, b, bmax[b], bsum[b], 10, dim, box5, bnt5, &conf5);
  level_box(sr, 64, 324, 18, a4x, a4y, b, bmax[b], bsum[b], 20, dim, box4, bnt4, &conf4);
  int mi = (conf4 > conf5) ? 1 : 0;
  const float* bb = mi ? box4 : box5;
  const float* bn = mi ? bnt4 : bnt5;
  for (int i = 0; i < 5; ++i) {
    boxes[b * 5 + i]   = bb[i];
    boxesNT[b * 5 + i] = bn[i];
  }
  rr[b] = fmaxf(0.0f, conf4 - conf5);
  __syncthreads();
  if (b == 0) {
    float s = 0.0f;
    for (int i = 0; i < 32; ++i) s += rr[i];
    reg[0] = s;
  }
}

// ---------------------------------------------------------------------------
// Host launcher
// ---------------------------------------------------------------------------
extern "C" void kernel_launch(void* const* d_in, const int* in_sizes, int n_in,
                              void* d_out, int out_size, void* d_ws, size_t ws_size,
                              hipStream_t stream) {
  (void)in_sizes; (void)n_in; (void)out_size; (void)ws_size;
  const float* x0     = (const float*)d_in[0];
  const float* x1     = (const float*)d_in[1];
  const float* stn0_w = (const float*)d_in[2];
  const float* stn0_b = (const float*)d_in[3];
  const float* stn1_w = (const float*)d_in[4];
  const float* conv_w = (const float*)d_in[5];
  const float* conv_b = (const float*)d_in[6];
  const float* last_w = (const float*)d_in[7];
  const float* last_b = (const float*)d_in[8];
  const int*   checkp = (const int*)d_in[9];
  const int*   dimp   = (const int*)d_in[10];
  float* out = (float*)d_out;

  const int NX0 = 32 * 256 * 100, NX1 = 32 * 256 * 400;
  const int NP0 = 32 * 100, NP1 = 32 * 400;          // conv pixel counts
  const int NW0 = 256 * 2304, NCW = 1024 * 2304, NLW = 201 * 1024;
  const int NLWP = 208 * 1024;                       // padded class weights
  const int N5 = 32 * 64, N4 = 32 * 324;             // sample rows

  char* ws = (char*)d_ws;
  size_t off = 0;
  auto take = [&](size_t bytes) -> char* {
    char* p = ws + off;
    off = (off + bytes + 255) & ~(size_t)255;
    return p;
  };
  h16*   w0h  = (h16*)take((size_t)NW0 * 2);
  h16*   cwh  = (h16*)take((size_t)NCW * 2);
  h16*   lwh  = (h16*)take((size_t)NLWP * 2);
  h16*   col0 = (h16*)take((size_t)NP0 * 2304 * 2);
  h16*   col1 = (h16*)take((size_t)NP1 * 2304 * 2);
  float* h0   = (float*)take((size_t)NX0 * 4);
  float* h1   = (float*)take((size_t)NX1 * 4);
  float* a5x  = (float*)take((size_t)N5 * 9 * 4);
  float* a5y  = (float*)take((size_t)N5 * 9 * 4);
  float* a4x  = (float*)take((size_t)N4 * 9 * 4);
  float* a4y  = (float*)take((size_t)N4 * 9 * 4);
  h16*   xs5  = (h16*)take((size_t)N5 * 2304 * 2);
  h16*   xs4  = (h16*)take((size_t)N4 * 2304 * 2);
  h16*   fr5  = (h16*)take((size_t)N5 * 1024 * 2);
  h16*   fr4  = (h16*)take((size_t)N4 * 1024 * 2);
  float* sbuf = (float*)take((size_t)32 * 201 * 388 * 4);
  float* bmax = (float*)take(32 * 4);
  float* bsum = (float*)take(32 * 4);

  // Output regions: likelihood | boxes | boxes_NT | theta_diff | reg
  float* lik   = out;            // 6432
  float* boxes = out + 6432;     // 160
  float* boxNT = out + 6592;     // 160
  float* tdiff = out + 6752;     // 74496
  float* reg   = out + 81248;    // 1

  // 1) f16 staging of weights (+ zero-pad last_w to 208 rows)
  cast_f16_kernel<<<(NW0 + 255) / 256, 256, 0, stream>>>(stn0_w, w0h, NW0, NW0);
  cast_f16_kernel<<<(NCW + 255) / 256, 256, 0, stream>>>(conv_w, cwh, NCW, NCW);
  cast_f16_kernel<<<(NLWP + 255) / 256, 256, 0, stream>>>(last_w, lwh, NLW, NLWP);

  // 2) im2col (f16) for both pyramid levels
  im2col_kernel<<<NP0, 256, 0, stream>>>(x0, col0, 10, 10, NP0);
  im2col_kernel<<<NP1, 256, 0, stream>>>(x1, col1, 20, 20, NP1);

  // 3) stn0 conv GEMM (WMMA): M=256, K=2304, N=3200/12800
  conv0_gemm_kernel<<<dim3(4, NP0 / 64), dim3(32, 4), 0, stream>>>(w0h, col0, stn0_b, h0, 10, 10);
  conv0_gemm_kernel<<<dim3(4, NP1 / 64), dim3(32, 4), 0, stream>>>(w0h, col1, stn0_b, h1, 20, 20);

  // 4) stn1 conv + theta + theta_diff + aux coords
  theta_aux_kernel<<<(N5 + 127) / 128, 128, 0, stream>>>(h0, stn1_w, checkp, tdiff, a5x, a5y, 8, 10, N5, 0);
  theta_aux_kernel<<<(N4 + 127) / 128, 128, 0, stream>>>(h1, stn1_w, checkp, tdiff, a4x, a4y, 18, 20, N4, N5);

  // 5) bilinear sampling -> xs f16
  sample_kernel<<<(N5 * 9 + 127) / 128, 128, 0, stream>>>(x0, a5x, a5y, xs5, 64, 10, N5 * 9);
  sample_kernel<<<(N4 * 9 + 127) / 128, 128, 0, stream>>>(x1, a4x, a4y, xs4, 324, 20, N4 * 9);

  // 6) feature GEMM (WMMA): rows x 2304 @ 2304 x 1024, bias+relu -> f16
  gemm_feat_kernel<<<dim3(N5 / 64, 16), dim3(32, 4), 0, stream>>>(xs5, cwh, conv_b, fr5);
  gemm_feat_kernel<<<dim3(N4 / 64, 16), dim3(32, 4), 0, stream>>>(xs4, cwh, conv_b, fr4);

  // 7) class GEMM (WMMA): 208 x 1024 @ 1024 x rows -> score buffer
  gemm_last_kernel<<<dim3(13, N5 / 128), dim3(32, 2), 0, stream>>>(lwh, fr5, last_b, sbuf, 8, 0);
  gemm_last_kernel<<<dim3(13, N4 / 128), dim3(32, 2), 0, stream>>>(lwh, fr4, last_b, sbuf, 18, 64);

  // 8) softmax stats + likelihood
  softmax_stats_kernel<<<32, 256, 0, stream>>>(sbuf, bmax, bsum);
  likelihood_kernel<<<(32 * 201 + 127) / 128, 128, 0, stream>>>(sbuf, bmax, bsum, lik);

  // 9) boxes / NT boxes / reg_score
  final_kernel<<<1, 32, 0, stream>>>(lik, sbuf, bmax, bsum, a5x, a5y, a4x, a4y,
                                     dimp, boxes, boxNT, reg);
}
